// Model_3461743640813
// MI455X (gfx1250) — matrix operand order repaired and checked by reading, not compiled, not run on hardware
//
#include <hip/hip_runtime.h>
#include <hip/hip_bf16.h>

// ---------------------------------------------------------------------------
// Persistent single-WGP bf16-WMMA LSTM for MI455X (gfx1250, wave32).
//   gates[64,512] = h[64,128] @ W_hh^T  (+ x*W_ih + bias), 4096 sequential steps
//   16 waves; wave w -> j = w&7 (H-col tile), m in {2*(w>>3), 2*(w>>3)+1}
//   W_hh kept in VGPRs as bf16 B-fragments; h double-buffered in LDS (bf16);
//   c state in VGPRs; one s_barrier per timestep.
// ---------------------------------------------------------------------------

typedef __attribute__((ext_vector_type(16))) __bf16       v16bf;
typedef __attribute__((ext_vector_type(8)))  float        v8f;
typedef __attribute__((ext_vector_type(4)))  unsigned int v4u;
typedef __attribute__((ext_vector_type(4)))  float        v4f;
typedef __attribute__((ext_vector_type(2)))  float        v2f;

union FragBF { v16bf v; v4u q[2]; };

#define B_ 64
#define H_ 128

__device__ __forceinline__ float fsig(float x) {
    float e = __expf(-x);
    return __builtin_amdgcn_rcpf(1.0f + e);
}
__device__ __forceinline__ float ftanh_(float x) {
    // tanh(x) = 2/(1+exp(-2x)) - 1
    float e = __expf(-2.0f * x);
    return 2.0f * __builtin_amdgcn_rcpf(1.0f + e) - 1.0f;
}

// data [B,T] -> x_tm [T,B] so the per-step x fetch is one 256B contiguous load
__global__ void transpose_data(const float* __restrict__ data,
                               float* __restrict__ x_tm, int T) {
    int o = blockIdx.x * blockDim.x + threadIdx.x;
    if (o < B_ * T) {
        int t = o >> 6;         // /64
        int b = o & 63;
        x_tm[o] = data[b * T + t];
    }
}

__global__ __launch_bounds__(512, 1) void lstm_persistent(
    const float* __restrict__ x_tm,  const float* __restrict__ h0,
    const float* __restrict__ c0,    const float* __restrict__ W_ih,
    const float* __restrict__ W_hh,  const float* __restrict__ b_ih,
    const float* __restrict__ b_hh,  const float* __restrict__ W_out,
    const float* __restrict__ b_out, float* __restrict__ out, int T) {

    __shared__ __align__(16) __bf16 hbuf[2][B_ * H_];   // 32 KB
    __shared__ __align__(16) float  xbuf[2][B_];        // 512 B

    const int tid  = threadIdx.x;
    const int wave = tid >> 5;
    const int lane = tid & 31;
    const int lh   = lane >> 4;      // lane half (0/1)
    const int n    = lane & 15;      // N column within tile (B/C/D layout)
    const int j    = wave & 7;       // H-column tile: cols 16j..16j+15
    const int mp   = wave >> 3;      // owns m = 2mp, 2mp+1

    // ---- one-time: W_hh^T bf16 B-fragments into registers -----------------
    // B[k][nn] = W_hh[gc0+nn][kbase+k]; layout: lane%16 = N, elems 0-7: K = lh*8..+7, elems 8-15: +16
    FragBF Bf[4][4];                 // [gate][kstep]
    #pragma unroll
    for (int g = 0; g < 4; ++g) {
        const int gc0 = g * H_ + j * 16;
        #pragma unroll
        for (int k = 0; k < 4; ++k) {
            const int kk = k * 32 + lh * 8;
            #pragma unroll
            for (int v = 0; v < 8; ++v) {
                float lo = W_hh[(gc0 + n) * H_ + kk + v];
                float hi = W_hh[(gc0 + n) * H_ + kk + 16 + v];
                Bf[g][k].v[v]     = (__bf16)lo;
                Bf[g][k].v[v + 8] = (__bf16)hi;
            }
        }
    }

    // per-lane input-projection scalars (rank-1 x path folded into bias FMA)
    float wih[4], bias[4];
    #pragma unroll
    for (int g = 0; g < 4; ++g) {
        int gc  = g * H_ + j * 16 + n;
        wih[g]  = W_ih[gc];
        bias[g] = b_ih[gc] + b_hh[gc];
    }

    // c state for the two owned tiles (C/D layout: row M = r + 8*lh)
    float cst[2][8];
    #pragma unroll
    for (int mi = 0; mi < 2; ++mi) {
        int m = 2 * mp + mi;
        #pragma unroll
        for (int r = 0; r < 8; ++r) {
            int b = m * 16 + lh * 8 + r;
            cst[mi][r] = c0[b * H_ + j * 16 + n];
        }
    }

    // stage h0 (bf16) and x[0]
    for (int i = tid; i < B_ * H_; i += 512) hbuf[0][i] = (__bf16)h0[i];
    if (tid < B_) xbuf[0][tid] = x_tm[tid];
    __syncthreads();

    int cur = 0;
    for (int t = 0; t < T; ++t) {
        const __bf16* hc = hbuf[cur];
        __bf16*       hn = hbuf[cur ^ 1];
        const float*  xc = xbuf[cur];

        #pragma unroll
        for (int mi = 0; mi < 2; ++mi) {
            const int m = 2 * mp + mi;

            // x values for owned rows: b = m*16 + lh*8 + r, r = 0..7
            const v4f* xp = (const v4f*)(xc + m * 16 + lh * 8);
            v4f xa0 = xp[0];
            v4f xa1 = xp[1];

            // seed accumulators: bias + x * W_ih
            v8f acc[4];
            #pragma unroll
            for (int g = 0; g < 4; ++g) {
                #pragma unroll
                for (int r = 0; r < 8; ++r) {
                    float xv = (r < 4) ? xa0[r & 3] : xa1[r & 3];
                    acc[g][r] = fmaf(xv, wih[g], bias[g]);
                }
            }

            // K loop: A fragment (lane = M, VGPRs 0-3: K = lh*8..+7,
            // VGPRs 4-7: +16) -> two ds_load_b128; then 4 WMMAs (i,f,g,o)
            #pragma unroll
            for (int k = 0; k < 4; ++k) {
                FragBF A;
                const v4u* p =
                    (const v4u*)(hc + ((m * 16 + n) * H_ + k * 32 + lh * 8));
                A.q[0] = p[0];
                A.q[1] = p[2];
                #pragma unroll
                for (int g = 0; g < 4; ++g) {
                    acc[g] = __builtin_amdgcn_wmma_f32_16x16x32_bf16(
                        false, A.v, false, Bf[g][k].v, (short)0, acc[g],
                        false, false);
                }
            }

            // LSTM cell update (gate order i,f,g,o) + bf16 h write
            #pragma unroll
            for (int r = 0; r < 8; ++r) {
                float iv = fsig(acc[0][r]);
                float fv = fsig(acc[1][r]);
                float gv = ftanh_(acc[2][r]);
                float ov = fsig(acc[3][r]);
                float c  = fmaf(fv, cst[mi][r], iv * gv);
                cst[mi][r] = c;
                float h  = ov * ftanh_(c);
                int   b  = m * 16 + lh * 8 + r;
                hn[b * H_ + j * 16 + n] = (__bf16)h;
            }
        }

        // software-pipelined x prefetch for step t+1 (wave 15, 256B total)
        if (wave == 15 && (t + 1) < T) {
            v2f xv = ((const v2f*)(x_tm + (t + 1) * B_))[lane];
            ((v2f*)xbuf[cur ^ 1])[lane] = xv;
        }

        __syncthreads();
        cur ^= 1;
    }

    // epilogue: out[b] = h_T . W_out + b_out   (64 tiny dot products)
    if (tid < B_) {
        const __bf16* hT = hbuf[cur];
        float s = b_out[0];
        #pragma unroll 4
        for (int k = 0; k < H_; ++k)
            s += (float)hT[tid * H_ + k] * W_out[k];
        out[tid] = s;
    }
}

extern "C" void kernel_launch(void* const* d_in, const int* in_sizes, int n_in,
                              void* d_out, int out_size, void* d_ws,
                              size_t ws_size, hipStream_t stream) {
    const float* data  = (const float*)d_in[0];
    const float* h0    = (const float*)d_in[1];
    const float* c0    = (const float*)d_in[2];
    const float* W_ih  = (const float*)d_in[3];
    const float* W_hh  = (const float*)d_in[4];
    const float* b_ih  = (const float*)d_in[5];
    const float* b_hh  = (const float*)d_in[6];
    const float* W_out = (const float*)d_in[7];
    const float* b_out = (const float*)d_in[8];
    float*       out   = (float*)d_out;

    const int T     = in_sizes[0] / B_;
    float*    x_tm  = (float*)d_ws;     // T*64 floats (1 MB for T=4096)
    const int total = B_ * T;

    transpose_data<<<(total + 255) / 256, 256, 0, stream>>>(data, x_tm, T);
    lstm_persistent<<<1, 512, 0, stream>>>(x_tm, h0, c0, W_ih, W_hh, b_ih,
                                           b_hh, W_out, b_out, out, T);
}
